// TransBlock_47708496724843
// MI455X (gfx1250) — matrix operand order repaired and checked by reading, not compiled, not run on hardware
//
#include <hip/hip_runtime.h>

// ---------------------------------------------------------------------------
// TransBlock: y = LN(x@W1+b1); attention(y); z = LN(o@W2+b2) + x
// B=8, C=2048, L=256, E=768, H=4, D=64, SCALE=8
// All GEMMs on v_wmma_f32_16x16x32_f16 (wave32), flash-attention inner loop.
// All fragments are 2x16B vector loads (global_load_b128 / ds_load_b128);
// K-loops unrolled so load clauses overlap WMMA issue.
// ---------------------------------------------------------------------------

typedef __attribute__((ext_vector_type(16))) _Float16 v16h;
typedef __attribute__((ext_vector_type(8)))  _Float16 v8h;
typedef __attribute__((ext_vector_type(8)))  float    v8f;

#define NBATCH 8
#define SEQ    2048
#define LFEAT  256
#define EMB    768
#define NHEAD  4
#define HDIM   64
#define NROWS  (NBATCH * SEQ)   // 16384
#define QKLD   512              // q|k packed row width

// workspace layout (bytes); total 48 MB
#define OFF_X16  ((size_t)0)                 //  8 MB : x   as f16 [16384,256]
#define OFF_W1T  ((size_t)8  << 20)          // 384 KB: W1^T as f16 [768,256]
#define OFF_W2T  (OFF_W1T + (size_t)EMB * LFEAT * 2)  // 128 KB: W2^T [256,256]
#define OFF_QK   ((size_t)16 << 20)          // 16 MB : q|k as f16 [16384,512]
#define OFF_VT   ((size_t)32 << 20)          //  8 MB : V^T as f16 [8,4,64,2048]
#define OFF_O    ((size_t)40 << 20)          //  8 MB : o   as f16 [16384,256]

static __device__ __forceinline__ v8f wmma_f16(v16h a, v16h b, v8f c) {
  return __builtin_amdgcn_wmma_f32_16x16x32_f16(
      /*neg_a=*/false, a, /*neg_b=*/false, b,
      /*c_mod=*/(short)0, c, /*reuse_a=*/false, /*reuse_b=*/false);
}

// two contiguous 16-byte loads -> one v16h fragment
static __device__ __forceinline__ v16h ld2x8(const _Float16* p0,
                                             const _Float16* p1) {
  union { v16h v; v8h h[2]; } u;
  u.h[0] = *(const v8h*)p0;
  u.h[1] = *(const v8h*)p1;
  return u.v;
}

// A fragment 16x32 f16 from row-major [*, ld] at (row0, k0).
// lane = m (lane&15); lane-group selects K 0..15/16..31.
static __device__ __forceinline__ v16h load_a16(const _Float16* __restrict__ base,
                                                int ld, int row0, int k0, int lane) {
  const int m = lane & 15, grp = lane >> 4;
  const _Float16* p = base + (size_t)(row0 + m) * ld + k0 + grp * 16;
  return ld2x8(p, p + 8);
}

// B fragment 32x16 f16 where the contraction dim K is CONTIGUOUS in memory:
// B[k][n] = base[(n0+n)*ld + k0+k].  lane = n; lane-group selects K 0..15/16..31.
// (used for K^T in QK^T, for N-major transposed weights, and for V^T)
static __device__ __forceinline__ v16h load_bk(const _Float16* __restrict__ base,
                                               int ld, int n0, int k0, int lane) {
  const int n = lane & 15, grp = lane >> 4;
  const _Float16* p = base + (size_t)(n0 + n) * ld + k0 + grp * 16;
  return ld2x8(p, p + 8);
}

// ---------------------------------------------------------------------------
__global__ __launch_bounds__(256) void cvt_f32_to_f16(const float* __restrict__ src,
                                                      _Float16* __restrict__ dst, int n) {
  int i = blockIdx.x * blockDim.x + threadIdx.x;
  if (i < n) dst[i] = (_Float16)src[i];
}

// src [K,N] fp32 row-major  ->  dst [N,K] f16 (N-major, K contiguous)
__global__ __launch_bounds__(256) void cvt_transpose_f16(const float* __restrict__ src,
                                                         _Float16* __restrict__ dst,
                                                         int K, int N) {
  int i = blockIdx.x * blockDim.x + threadIdx.x;
  if (i < K * N) {
    int k = i / N, n = i - k * N;
    dst[(size_t)n * K + k] = (_Float16)src[i];
  }
}

// ---------------------------------------------------------------------------
// Kernel 1: y = LN(x @ W1 + b1); write q|k row-major [16384,512] and V
// transposed [B,H,64,2048].
// block = 16 rows x 768 cols, 8 waves, 6 (16x16) tiles / wave, K = 256 (8x32)
// ---------------------------------------------------------------------------
__global__ __launch_bounds__(256) void gemm1_ln_kernel(
    const _Float16* __restrict__ x16, const _Float16* __restrict__ w1t,
    const float* __restrict__ b1, const float* __restrict__ g1,
    const float* __restrict__ beta1, _Float16* __restrict__ qk16,
    _Float16* __restrict__ vt16) {
  __shared__ float rs[16], rq[16];
  const int lane = threadIdx.x & 31;
  const int wave = threadIdx.x >> 5;
  const int row0 = blockIdx.x * 16;
  const int n0   = wave * 96;              // 6 tiles of 16 columns
  const int grp  = lane >> 4, nl = lane & 15;

  if (threadIdx.x < 16) { rs[threadIdx.x] = 0.f; rq[threadIdx.x] = 0.f; }
  __syncthreads();

  v8f acc[6] = {};
  // unroll 2: lets the scheduler clause-issue the next iteration's loads
  // while the current iteration's WMMAs execute
#pragma unroll 2
  for (int kb = 0; kb < 8; ++kb) {
    v16h a = load_a16(x16, LFEAT, row0, kb * 32, lane);
    v16h b0 = load_bk(w1t, LFEAT, n0 + 0 * 16, kb * 32, lane);
    v16h b1f = load_bk(w1t, LFEAT, n0 + 1 * 16, kb * 32, lane);
    v16h b2 = load_bk(w1t, LFEAT, n0 + 2 * 16, kb * 32, lane);
    v16h b3 = load_bk(w1t, LFEAT, n0 + 3 * 16, kb * 32, lane);
    v16h b4 = load_bk(w1t, LFEAT, n0 + 4 * 16, kb * 32, lane);
    v16h b5 = load_bk(w1t, LFEAT, n0 + 5 * 16, kb * 32, lane);
    acc[0] = wmma_f16(a, b0, acc[0]);
    acc[1] = wmma_f16(a, b1f, acc[1]);
    acc[2] = wmma_f16(a, b2, acc[2]);
    acc[3] = wmma_f16(a, b3, acc[3]);
    acc[4] = wmma_f16(a, b4, acc[4]);
    acc[5] = wmma_f16(a, b5, acc[5]);
  }

  // bias
#pragma unroll
  for (int t = 0; t < 6; ++t) {
    const float bias = b1[n0 + t * 16 + nl];
#pragma unroll
    for (int g = 0; g < 8; ++g) acc[t][g] += bias;
  }

  // row statistics: in-group shuffle reduce, then one LDS atomic per row/wave
#pragma unroll
  for (int g = 0; g < 8; ++g) {
    float s = 0.f, q = 0.f;
#pragma unroll
    for (int t = 0; t < 6; ++t) { float v = acc[t][g]; s += v; q += v * v; }
#pragma unroll
    for (int mask = 1; mask < 16; mask <<= 1) {
      s += __shfl_xor(s, mask, 32);
      q += __shfl_xor(q, mask, 32);
    }
    if (nl == 0) {
      atomicAdd(&rs[g + 8 * grp], s);
      atomicAdd(&rq[g + 8 * grp], q);
    }
  }
  __syncthreads();

#pragma unroll
  for (int g = 0; g < 8; ++g) {
    const int m = g + 8 * grp;
    const int row = row0 + m;
    const float mu   = rs[m] * (1.f / EMB);
    const float var  = rq[m] * (1.f / EMB) - mu * mu;
    const float rstd = rsqrtf(var + 1e-5f);
#pragma unroll
    for (int t = 0; t < 6; ++t) {
      const int ncol = n0 + t * 16 + nl;
      const float v = (acc[t][g] - mu) * rstd * g1[ncol] + beta1[ncol];
      if (ncol < 2 * LFEAT) {
        // q | k region, row-major ld=512
        qk16[(size_t)row * QKLD + ncol] = (_Float16)v;
      } else {
        // v region: store transposed  vt[b][h][d][c]
        const int cc = ncol - 2 * LFEAT;     // h*64 + d
        const int bb = row >> 11, c = row & (SEQ - 1);
        vt16[((size_t)bb * NHEAD * HDIM + cc) * SEQ + c] = (_Float16)v;
      }
    }
  }
}

// ---------------------------------------------------------------------------
// Kernel 2: flash attention per (batch, head).
// block = 128 queries (8 waves x 16 rows); stream keys in chunks of 32.
// q at qk cols [h*64), k at [256+h*64); V from transposed vt buffer.
// ---------------------------------------------------------------------------
__global__ __launch_bounds__(256) void attn_kernel(const _Float16* __restrict__ qk16,
                                                   const _Float16* __restrict__ vt16,
                                                   _Float16* __restrict__ o16) {
  __shared__ _Float16 Plds[8][16 * 32];    // P staging, 1 KB per wave
  const int lane = threadIdx.x & 31;
  const int wave = threadIdx.x >> 5;
  const int bh = blockIdx.y;
  const int b = bh >> 2, h = bh & 3;
  const int q0 = blockIdx.x * 128 + wave * 16;
  const int grp = lane >> 4, nl = lane & 15;

  const _Float16* qp = qk16 + (size_t)b * SEQ * QKLD + h * HDIM;
  const _Float16* kp = qp + LFEAT;                       // k cols start at 256
  const _Float16* vtp = vt16 + (size_t)bh * HDIM * SEQ;  // [64, 2048] K-contig
  _Float16* P = &Plds[wave][0];

  v8f o[4] = {};
  float mi[8], li[8];
#pragma unroll
  for (int g = 0; g < 8; ++g) { mi[g] = -1e30f; li[g] = 0.f; }

  const v16h qa0 = load_a16(qp, QKLD, q0, 0, lane);
  const v16h qa1 = load_a16(qp, QKLD, q0, 32, lane);

#pragma unroll 1
  for (int kb = 0; kb < SEQ; kb += 32) {
    if (kb + 32 < SEQ) {       // prefetch next key block (K rows + V^T run)
      __builtin_prefetch(kp + (size_t)(kb + 32) * QKLD, 0, 3);
      __builtin_prefetch(vtp + kb + 32, 0, 3);
    }
    // S = Q . K^T for 32 keys (two 16x16 tiles), contraction over 64 dims
    v8f s0 = {}, s1 = {};
    s0 = wmma_f16(qa0, load_bk(kp, QKLD, kb, 0, lane), s0);
    s0 = wmma_f16(qa1, load_bk(kp, QKLD, kb, 32, lane), s0);
    s1 = wmma_f16(qa0, load_bk(kp, QKLD, kb + 16, 0, lane), s1);
    s1 = wmma_f16(qa1, load_bk(kp, QKLD, kb + 16, 32, lane), s1);

    // issue V-fragment loads NOW so their latency hides behind the softmax VALU
    v16h bv0 = load_bk(vtp, SEQ, 0 * 16, kb, lane);
    v16h bv1 = load_bk(vtp, SEQ, 1 * 16, kb, lane);
    v16h bv2 = load_bk(vtp, SEQ, 2 * 16, kb, lane);
    v16h bv3 = load_bk(vtp, SEQ, 3 * 16, kb, lane);

    // online softmax (per row: shuffle reduce across the 16-lane group)
#pragma unroll
    for (int g = 0; g < 8; ++g) {
      float a0 = s0[g] * 0.125f;             // /SCALE
      float a1 = s1[g] * 0.125f;
      float mx = fmaxf(a0, a1);
#pragma unroll
      for (int mask = 1; mask < 16; mask <<= 1)
        mx = fmaxf(mx, __shfl_xor(mx, mask, 32));
      const float mnew = fmaxf(mi[g], mx);
      const float corr = __expf(mi[g] - mnew);
      mi[g] = mnew;
      const float p0 = __expf(a0 - mnew);
      const float p1 = __expf(a1 - mnew);
      float ps = p0 + p1;
#pragma unroll
      for (int mask = 1; mask < 16; mask <<= 1)
        ps += __shfl_xor(ps, mask, 32);
      li[g] = li[g] * corr + ps;
      const int m = g + 8 * grp;
      P[m * 32 + nl]      = (_Float16)p0;    // re-layout P via LDS for A frag
      P[m * 32 + 16 + nl] = (_Float16)p1;
#pragma unroll
      for (int t = 0; t < 4; ++t) o[t][g] *= corr;
    }

    // O += P . V   (contraction over the 32 keys; 4 tiles cover 64 dims)
    const v16h pa = load_a16(P, 32, 0, 0, lane);   // 2x ds_load_b128
    o[0] = wmma_f16(pa, bv0, o[0]);
    o[1] = wmma_f16(pa, bv1, o[1]);
    o[2] = wmma_f16(pa, bv2, o[2]);
    o[3] = wmma_f16(pa, bv3, o[3]);
  }

#pragma unroll
  for (int g = 0; g < 8; ++g) {
    const int m = g + 8 * grp;
    const float inv = 1.f / li[g];
    const size_t row = (size_t)b * SEQ + q0 + m;
#pragma unroll
    for (int t = 0; t < 4; ++t)
      o16[row * LFEAT + h * HDIM + t * 16 + nl] = (_Float16)(o[t][g] * inv);
  }
}

// ---------------------------------------------------------------------------
// Kernel 3: z = LN(o @ W2 + b2) + x   -> out fp32 [16384, 256]
// block = 16 rows x 256 cols, 8 waves, 2 tiles / wave; K-loop fully unrolled
// ---------------------------------------------------------------------------
__global__ __launch_bounds__(256) void gemm2_ln_res_kernel(
    const _Float16* __restrict__ o16, const _Float16* __restrict__ w2t,
    const float* __restrict__ b2, const float* __restrict__ g2,
    const float* __restrict__ beta2, const float* __restrict__ x,
    float* __restrict__ out) {
  __shared__ float rs[16], rq[16];
  const int lane = threadIdx.x & 31;
  const int wave = threadIdx.x >> 5;
  const int row0 = blockIdx.x * 16;
  const int n0   = wave * 32;              // 2 tiles of 16 columns
  const int grp  = lane >> 4, nl = lane & 15;

  if (threadIdx.x < 16) { rs[threadIdx.x] = 0.f; rq[threadIdx.x] = 0.f; }
  __syncthreads();

  v8f acc[2] = {};
#pragma unroll
  for (int kb = 0; kb < 8; ++kb) {
    v16h a = load_a16(o16, LFEAT, row0, kb * 32, lane);
    v16h b0 = load_bk(w2t, LFEAT, n0 + 0 * 16, kb * 32, lane);
    v16h b1f = load_bk(w2t, LFEAT, n0 + 1 * 16, kb * 32, lane);
    acc[0] = wmma_f16(a, b0, acc[0]);
    acc[1] = wmma_f16(a, b1f, acc[1]);
  }

#pragma unroll
  for (int t = 0; t < 2; ++t) {
    const float bias = b2[n0 + t * 16 + nl];
#pragma unroll
    for (int g = 0; g < 8; ++g) acc[t][g] += bias;
  }

#pragma unroll
  for (int g = 0; g < 8; ++g) {
    float s = 0.f, q = 0.f;
#pragma unroll
    for (int t = 0; t < 2; ++t) { float v = acc[t][g]; s += v; q += v * v; }
#pragma unroll
    for (int mask = 1; mask < 16; mask <<= 1) {
      s += __shfl_xor(s, mask, 32);
      q += __shfl_xor(q, mask, 32);
    }
    if (nl == 0) {
      atomicAdd(&rs[g + 8 * grp], s);
      atomicAdd(&rq[g + 8 * grp], q);
    }
  }
  __syncthreads();

#pragma unroll
  for (int g = 0; g < 8; ++g) {
    const int m = g + 8 * grp;
    const float mu   = rs[m] * (1.f / LFEAT);
    const float var  = rq[m] * (1.f / LFEAT) - mu * mu;
    const float rstd = rsqrtf(var + 1e-5f);
#pragma unroll
    for (int t = 0; t < 2; ++t) {
      const int ncol = n0 + t * 16 + nl;
      const size_t idx = (size_t)(row0 + m) * LFEAT + ncol;
      out[idx] = (acc[t][g] - mu) * rstd * g2[ncol] + beta2[ncol] + x[idx];
    }
  }
}

// ---------------------------------------------------------------------------
extern "C" void kernel_launch(void* const* d_in, const int* in_sizes, int n_in,
                              void* d_out, int out_size, void* d_ws, size_t ws_size,
                              hipStream_t stream) {
  (void)in_sizes; (void)n_in; (void)out_size; (void)ws_size;

  const float* x     = (const float*)d_in[0];
  const float* W1    = (const float*)d_in[1];
  const float* b1    = (const float*)d_in[2];
  const float* g1    = (const float*)d_in[3];
  const float* beta1 = (const float*)d_in[4];
  const float* W2    = (const float*)d_in[5];
  const float* b2    = (const float*)d_in[6];
  const float* g2    = (const float*)d_in[7];
  const float* beta2 = (const float*)d_in[8];
  float* out = (float*)d_out;

  char* ws = (char*)d_ws;
  _Float16* x16  = (_Float16*)(ws + OFF_X16);
  _Float16* w1t  = (_Float16*)(ws + OFF_W1T);
  _Float16* w2t  = (_Float16*)(ws + OFF_W2T);
  _Float16* qk16 = (_Float16*)(ws + OFF_QK);
  _Float16* vt16 = (_Float16*)(ws + OFF_VT);
  _Float16* o16  = (_Float16*)(ws + OFF_O);

  const int nx  = NROWS * LFEAT;   // 4194304
  const int nw1 = LFEAT * EMB;     // 196608
  const int nw2 = LFEAT * LFEAT;   // 65536
  cvt_f32_to_f16<<<(nx + 255) / 256, 256, 0, stream>>>(x, x16, nx);
  cvt_transpose_f16<<<(nw1 + 255) / 256, 256, 0, stream>>>(W1, w1t, LFEAT, EMB);
  cvt_transpose_f16<<<(nw2 + 255) / 256, 256, 0, stream>>>(W2, w2t, LFEAT, LFEAT);

  gemm1_ln_kernel<<<NROWS / 16, 256, 0, stream>>>(x16, w1t, b1, g1, beta1,
                                                  qk16, vt16);
  attn_kernel<<<dim3(SEQ / 128, NBATCH * NHEAD), 256, 0, stream>>>(qk16, vt16, o16);
  gemm2_ln_res_kernel<<<NROWS / 16, 256, 0, stream>>>(o16, w2t, b2, g2, beta2,
                                                      x, out);
}
